// RelMultiHeadSelfAttention_45543833207175
// MI455X (gfx1250) — matrix operand order repaired and checked by reading, not compiled, not run on hardware
//
#include <hip/hip_runtime.h>

// ---------------------------------------------------------------------------
// Transformer-XL relative multi-head attention for MI455X (gfx1250).
// All GEMMs via v_wmma_f32_16x16x32_bf16 (wave32, f32 accumulate).
// f32 -> bf16 conversion happens exactly once per tensor; all fragment loads
// are b128 (two 16B loads per fragment per lane).
// ---------------------------------------------------------------------------

typedef __attribute__((ext_vector_type(16))) __bf16 v16bf;
typedef __attribute__((ext_vector_type(8)))  float  v8f;

#define B_   2
#define H_   16
#define T_   1024
#define MM_  1024
#define L_   2048
#define J_   2048
#define DH_  64
#define D_   1024
#define SCALE_ 0.125f   // 1/sqrt(64)

union FragU {
  v16bf bf;
  unsigned short us[16];
  unsigned int   w[8];
  uint4          q[2];
};

__device__ __forceinline__ unsigned short f2bf(float f) {
  unsigned u = __float_as_uint(f);
  u += 0x7FFFu + ((u >> 16) & 1u);   // round-to-nearest-even
  return (unsigned short)(u >> 16);
}
__device__ __forceinline__ float bf2f(unsigned short h) {
  return __uint_as_float(((unsigned)h) << 16);
}
// A-operand K pair base for index math on LDS probabilities (ISA 7.12.2)
__device__ __forceinline__ int kA(int v, int l) {
  return ((v & 3) * 2) + ((v & 4) ? 16 : 0) + ((l & 16) ? 8 : 0);
}

__device__ __forceinline__ v8f wmma_bf16(v16bf a, v16bf b, v8f c) {
  return __builtin_amdgcn_wmma_f32_16x16x32_bf16(
      false, a, false, b, (short)0, c, false, false);
}
__device__ __forceinline__ v8f zero8() {
  v8f z;
#pragma unroll
  for (int i = 0; i < 8; ++i) z[i] = 0.0f;
  return z;
}

// ---- b128 fragment loaders (bf16 operands) --------------------------------
// A-matrix 16x32: lane -> M=lane&15; lane-local data = elems [k0+s, k0+s+8)
// and [k0+s+16, k0+s+24) where s = 8*(lane>=16). Two 16B loads.
__device__ __forceinline__ v16bf fragA_bf16_row(const unsigned short* row, int k0) {
  const int l = (int)(threadIdx.x & 31);
  const unsigned short* rp = row + k0 + ((l & 16) ? 8 : 0);
  FragU u;
  u.q[0] = *(const uint4*)(rp);        // K pairs 0..3  (elems +0..7)
  u.q[1] = *(const uint4*)(rp + 16);   // K pairs 4..7  (elems +16..23)
  return u.bf;
}
__device__ __forceinline__ v16bf fragA_bf16(const unsigned short* base, int ld,
                                            int m0, int k0) {
  const int l = (int)(threadIdx.x & 31);
  return fragA_bf16_row(base + (size_t)(m0 + (l & 15)) * ld, k0);
}
// B-matrix 32x16: lane -> N=lane&15; lane-local data = elems [k0+s, k0+s+8)
// and [k0+s+16, k0+s+24) where s = 8*(lane>=16). Two 16B loads.
__device__ __forceinline__ v16bf fragB_bf16(const unsigned short* base, int ld,
                                            int n0, int k0) {
  const int l = (int)(threadIdx.x & 31);
  const unsigned short* rp =
      base + (size_t)(n0 + (l & 15)) * ld + k0 + ((l & 16) ? 8 : 0);
  FragU u;
  u.q[0] = *(const uint4*)(rp);        // K pairs 0..3
  u.q[1] = *(const uint4*)(rp + 16);   // K pairs 4..7
  return u.bf;
}

// ---------------------------------------------------------------------------
// Kernel 0: f32 -> bf16 bulk convert (8 elems / thread, b128 in & out)
// ---------------------------------------------------------------------------
__global__ void __launch_bounds__(256) k_cvt(const float* __restrict__ src,
                                             unsigned short* __restrict__ dst,
                                             int n8) {
  const int i = blockIdx.x * 256 + threadIdx.x;
  if (i >= n8) return;
  const float4* s = (const float4*)src + (size_t)i * 2;
  const float4 a = s[0], b = s[1];
  uint4 o;
  o.x = (unsigned)f2bf(a.x) | ((unsigned)f2bf(a.y) << 16);
  o.y = (unsigned)f2bf(a.z) | ((unsigned)f2bf(a.w) << 16);
  o.z = (unsigned)f2bf(b.x) | ((unsigned)f2bf(b.y) << 16);
  o.w = (unsigned)f2bf(b.z) | ((unsigned)f2bf(b.w) << 16);
  ((uint4*)dst)[i] = o;
}

// ---------------------------------------------------------------------------
// Kernel 1: q projection; writes q+pos_bias_u and q+pos_bias_v, bf16 [b,h,t,d]
// ---------------------------------------------------------------------------
__global__ void __launch_bounds__(256) k_proj_q(
    const unsigned short* __restrict__ xb, const unsigned short* __restrict__ qwb,
    const float* __restrict__ bu, const float* __restrict__ bv,
    unsigned short* __restrict__ qu, unsigned short* __restrict__ qv) {
  const int NT = D_ / 16;
  const int tile = blockIdx.x * 8 + (threadIdx.x >> 5);
  if (tile >= (B_ * T_ / 16) * NT) return;
  const int tm = tile / NT, tn = tile % NT;
  const int l = threadIdx.x & 31;
  const int gm = tm * 16, gn = tn * 16;

  v8f acc = zero8();
#pragma unroll 4
  for (int k0 = 0; k0 < D_; k0 += 32) {
    v16bf a = fragA_bf16(xb,  D_, gm, k0);
    v16bf b = fragB_bf16(qwb, D_, gn, k0);
    acc = wmma_bf16(a, b, acc);
  }

  const int n = gn + (l & 15);
  const int h = n >> 6, d = n & 63;
  const float ub = bu[n], vb = bv[n];
#pragma unroll
  for (int v = 0; v < 8; ++v) {
    const int m = gm + v + ((l & 16) ? 8 : 0);
    const int bb = m >> 10, t = m & (T_ - 1);
    const size_t o = ((size_t)((bb * H_ + h) * T_ + t)) * DH_ + d;
    qu[o] = f2bf(acc[v] + ub);
    qv[o] = f2bf(acc[v] + vb);
  }
}

// ---------------------------------------------------------------------------
// Kernel 2: k & v projections of cat(memory, x) (cat is virtual).
// k -> bf16 [b,h,L,d];  v -> bf16 transposed [b,h,d,L]
// ---------------------------------------------------------------------------
__global__ void __launch_bounds__(256) k_proj_kv(
    const unsigned short* __restrict__ xb, const unsigned short* __restrict__ memb,
    const unsigned short* __restrict__ kwb, const unsigned short* __restrict__ vwb,
    unsigned short* __restrict__ kk, unsigned short* __restrict__ vt) {
  const int NT = D_ / 16;
  const int tile = blockIdx.x * 8 + (threadIdx.x >> 5);
  if (tile >= (B_ * L_ / 16) * NT) return;
  const int tm = tile / NT, tn = tile % NT;
  const int l = threadIdx.x & 31;

  const int mrow = tm * 16 + (l & 15);
  const int bb0 = mrow >> 11, pos0 = mrow & (L_ - 1);
  const unsigned short* arow = (pos0 < MM_)
      ? (memb + ((size_t)(bb0 * MM_ + pos0)) * D_)
      : (xb   + ((size_t)(bb0 * T_ + pos0 - MM_)) * D_);

  v8f ak = zero8(), av = zero8();
#pragma unroll 4
  for (int k0 = 0; k0 < D_; k0 += 32) {
    v16bf a   = fragA_bf16_row(arow, k0);
    v16bf bk  = fragB_bf16(kwb, D_, tn * 16, k0);
    v16bf bv2 = fragB_bf16(vwb, D_, tn * 16, k0);
    ak = wmma_bf16(a, bk,  ak);
    av = wmma_bf16(a, bv2, av);
  }

  const int n = tn * 16 + (l & 15);
  const int h = n >> 6, d = n & 63;
#pragma unroll
  for (int v = 0; v < 8; ++v) {
    const int m = tm * 16 + v + ((l & 16) ? 8 : 0);
    const int bb = m >> 11, p = m & (L_ - 1);
    kk[((size_t)(bb * H_ + h) * L_ + p) * DH_ + d] = f2bf(ak[v]);
    vt[(((size_t)(bb * H_ + h)) * DH_ + d) * L_ + p] = f2bf(av[v]);
  }
}

// ---------------------------------------------------------------------------
// Kernel 3: r projection: pos_emb @ r_w.T -> bf16 [h, J, d]
// ---------------------------------------------------------------------------
__global__ void __launch_bounds__(256) k_proj_r(
    const unsigned short* __restrict__ peb, const unsigned short* __restrict__ rwb,
    unsigned short* __restrict__ rr) {
  const int NT = D_ / 16;
  const int tile = blockIdx.x * 8 + (threadIdx.x >> 5);
  if (tile >= (J_ / 16) * NT) return;
  const int tm = tile / NT, tn = tile % NT;
  const int l = threadIdx.x & 31;

  v8f acc = zero8();
#pragma unroll 4
  for (int k0 = 0; k0 < D_; k0 += 32) {
    v16bf a = fragA_bf16(peb, D_, tm * 16, k0);
    v16bf b = fragB_bf16(rwb, D_, tn * 16, k0);
    acc = wmma_bf16(a, b, acc);
  }
  const int n = tn * 16 + (l & 15);
  const int h = n >> 6, d = n & 63;
#pragma unroll
  for (int v = 0; v < 8; ++v) {
    const int j = tm * 16 + v + ((l & 16) ? 8 : 0);
    rr[((size_t)(h * J_ + j)) * DH_ + d] = f2bf(acc[v]);
  }
}

// ---------------------------------------------------------------------------
// Kernel 4: fused rel-shift flash attention.
// One block = (b, h, 16 query rows). 8 waves split the 2048 keys.
// Phase 1: rel_pre[16 x 2048] = (q+v_bias) @ r^T via WMMA, staged in LDS bf16,
//          plus scalar row i0+16 for the rel-shift wrap.
// Phase 2: streaming softmax of (content + shifted-rel)*scale, O += P @ V.
// ---------------------------------------------------------------------------
#define RELS 2080            // padded LDS row stride (bf16 elements)

extern __shared__ char smem[];

__global__ void __launch_bounds__(256, 1) k_attn(
    const unsigned short* __restrict__ qu, const unsigned short* __restrict__ qv,
    const unsigned short* __restrict__ kk, const unsigned short* __restrict__ vt,
    const unsigned short* __restrict__ rr, unsigned short* __restrict__ attn_o) {
  unsigned short* relS  = (unsigned short*)smem;                      // 16*RELS bf16
  float*          rel17 = (float*)(smem + 16 * RELS * 2);             // 1024 f32
  unsigned short* pstg  = (unsigned short*)(smem + 16 * RELS * 2 + 4096); // 8*512 bf16
  float* mstat = (float*)(smem + 16 * RELS * 2 + 4096 + 8192);        // 8*16
  float* lstat = mstat + 128;                                         // 8*16
  float* Oc    = lstat + 128;                                         // 8*16*64

  const int bid = blockIdx.x;
  const int qi = bid & 63;           // T_/16 == 64
  const int bh = bid >> 6;
  const int h = bh & (H_ - 1), b = bh >> 4;
  const int i0 = qi * 16;
  const int tid = threadIdx.x, w = tid >> 5, l = tid & 31;
  const int nl = l & 15, hi = (l >> 4) & 1;

  const unsigned short* quB = qu + ((size_t)(b * H_ + h)) * T_ * DH_;
  const unsigned short* qvB = qv + ((size_t)(b * H_ + h)) * T_ * DH_;
  const unsigned short* kkB = kk + ((size_t)(b * H_ + h)) * L_ * DH_;
  const unsigned short* vtB = vt + ((size_t)(b * H_ + h)) * DH_ * L_;
  const unsigned short* rrB = rr + ((size_t)h) * J_ * DH_;

  // ---- phase 1: rel_pre row-block into LDS ----
  {
    v16bf aqv0 = fragA_bf16(qvB, DH_, i0, 0);
    v16bf aqv1 = fragA_bf16(qvB, DH_, i0, 32);
#pragma unroll 2
    for (int t = w; t < J_ / 16; t += 8) {
      const int p0 = t * 16;
      v8f acc = zero8();
      acc = wmma_bf16(aqv0, fragB_bf16(rrB, DH_, p0, 0),  acc);
      acc = wmma_bf16(aqv1, fragB_bf16(rrB, DH_, p0, 32), acc);
#pragma unroll
      for (int v = 0; v < 8; ++v) {
        const int mloc = v + hi * 8;
        relS[mloc * RELS + p0 + nl] = f2bf(acc[v]);
      }
    }
  }
  if (i0 + 16 < T_) {        // wrap row for i = i0+16 (small p' only)
    const unsigned short* qrow = qvB + (size_t)(i0 + 16) * DH_;
    for (int p = tid; p < 1024; p += 256) {
      const unsigned short* rrow = rrB + (size_t)p * DH_;
      float s = 0.0f;
#pragma unroll
      for (int d = 0; d < DH_; ++d) s += bf2f(qrow[d]) * bf2f(rrow[d]);
      rel17[p] = s;
    }
  }
  __syncthreads();

  // ---- phase 2: flash loop, wave w owns keys [w*256, w*256+256) ----
  v16bf aqu0 = fragA_bf16(quB, DH_, i0, 0);
  v16bf aqu1 = fragA_bf16(quB, DH_, i0, 32);
  float mrun[8], lrun[8];
#pragma unroll
  for (int v = 0; v < 8; ++v) { mrun[v] = -1.0e30f; lrun[v] = 0.0f; }
  v8f O0 = zero8(), O1 = zero8(), O2 = zero8(), O3 = zero8();
  unsigned short* pw = pstg + w * 512;

#pragma unroll 2
  for (int ch = 0; ch < 8; ++ch) {
    const int j0 = w * 256 + ch * 32;

    v8f s0 = zero8(), s1 = zero8();
    s0 = wmma_bf16(aqu0, fragB_bf16(kkB, DH_, j0,      0),  s0);
    s0 = wmma_bf16(aqu1, fragB_bf16(kkB, DH_, j0,      32), s0);
    s1 = wmma_bf16(aqu0, fragB_bf16(kkB, DH_, j0 + 16, 0),  s1);
    s1 = wmma_bf16(aqu1, fragB_bf16(kkB, DH_, j0 + 16, 32), s1);

    // add the rel-shifted bias and scale
#pragma unroll
    for (int v = 0; v < 8; ++v) {
      const int mloc = v + hi * 8;
      const int i = i0 + mloc;
#pragma unroll
      for (int t = 0; t < 2; ++t) {
        const int j = j0 + t * 16 + nl;
        const int dd = j - i;
        float rv;
        if (dd <= MM_) {
          rv = bf2f(relS[mloc * RELS + dd + (T_ - 1)]);
        } else if (dd == MM_ + 1) {
          rv = 0.0f;
        } else {
          const int pp = dd - MM_ - 2;
          rv = (mloc < 15) ? bf2f(relS[(mloc + 1) * RELS + pp]) : rel17[pp];
        }
        const float sv = ((t ? s1[v] : s0[v]) + rv) * SCALE_;
        if (t) s1[v] = sv; else s0[v] = sv;
      }
    }

    // streaming softmax update, row groups of 16 lanes (wave32)
#pragma unroll
    for (int v = 0; v < 8; ++v) {
      float mx = fmaxf(s0[v], s1[v]);
      mx = fmaxf(mx, __shfl_xor(mx, 8, 32));
      mx = fmaxf(mx, __shfl_xor(mx, 4, 32));
      mx = fmaxf(mx, __shfl_xor(mx, 2, 32));
      mx = fmaxf(mx, __shfl_xor(mx, 1, 32));
      const float mnew  = fmaxf(mrun[v], mx);
      const float alpha = __expf(mrun[v] - mnew);
      const float p0v = __expf(s0[v] - mnew);
      const float p1v = __expf(s1[v] - mnew);
      float rs = p0v + p1v;
      rs += __shfl_xor(rs, 8, 32);
      rs += __shfl_xor(rs, 4, 32);
      rs += __shfl_xor(rs, 2, 32);
      rs += __shfl_xor(rs, 1, 32);
      lrun[v] = lrun[v] * alpha + rs;
      mrun[v] = mnew;
      O0[v] *= alpha; O1[v] *= alpha; O2[v] *= alpha; O3[v] *= alpha;
      const int mloc = v + hi * 8;
      pw[mloc * 32 + nl]      = f2bf(p0v);   // P tile 16x32 in wave-private LDS
      pw[mloc * 32 + 16 + nl] = f2bf(p1v);
    }

    // re-layout P: C-layout -> A-operand via LDS round trip (same wave,
    // DS ops stay in order; b128 reloads)
    const v16bf fp = fragA_bf16_row(pw + (size_t)nl * 32, 0);

    O0 = wmma_bf16(fp, fragB_bf16(vtB, L_, 0,  j0), O0);
    O1 = wmma_bf16(fp, fragB_bf16(vtB, L_, 16, j0), O1);
    O2 = wmma_bf16(fp, fragB_bf16(vtB, L_, 32, j0), O2);
    O3 = wmma_bf16(fp, fragB_bf16(vtB, L_, 48, j0), O3);
  }

  // ---- cross-wave combine through LDS ----
  if (nl == 0) {
#pragma unroll
    for (int v = 0; v < 8; ++v) {
      const int row = v + hi * 8;
      mstat[w * 16 + row] = mrun[v];
      lstat[w * 16 + row] = lrun[v];
    }
  }
#pragma unroll
  for (int v = 0; v < 8; ++v) {
    const int row = v + hi * 8;
    float* od = Oc + (size_t)(w * 16 + row) * 64;
    od[0  + nl] = O0[v];
    od[16 + nl] = O1[v];
    od[32 + nl] = O2[v];
    od[48 + nl] = O3[v];
  }
  __syncthreads();

  for (int idx = tid; idx < 1024; idx += 256) {
    const int row = idx >> 6, d = idx & 63;
    float gm = -1.0e30f;
#pragma unroll
    for (int ww = 0; ww < 8; ++ww) gm = fmaxf(gm, mstat[ww * 16 + row]);
    float ls = 0.0f, os = 0.0f;
#pragma unroll
    for (int ww = 0; ww < 8; ++ww) {
      const float e = __expf(mstat[ww * 16 + row] - gm);
      ls += lstat[ww * 16 + row] * e;
      os += Oc[(size_t)(ww * 16 + row) * 64 + d] * e;
    }
    attn_o[((size_t)(b * T_ + i0 + row)) * D_ + h * DH_ + d] = f2bf(os / ls);
  }
}

// ---------------------------------------------------------------------------
// Kernel 5: output projection: attn(bf16) @ out_w.T(bf16) -> f32 d_out
// ---------------------------------------------------------------------------
__global__ void __launch_bounds__(256) k_out(
    const unsigned short* __restrict__ attn, const unsigned short* __restrict__ owb,
    float* __restrict__ out) {
  const int NT = D_ / 16;
  const int tile = blockIdx.x * 8 + (threadIdx.x >> 5);
  if (tile >= (B_ * T_ / 16) * NT) return;
  const int tm = tile / NT, tn = tile % NT;
  const int l = threadIdx.x & 31;

  v8f acc = zero8();
#pragma unroll 4
  for (int k0 = 0; k0 < D_; k0 += 32) {
    v16bf a = fragA_bf16(attn, D_, tm * 16, k0);
    v16bf b = fragB_bf16(owb, D_, tn * 16, k0);
    acc = wmma_bf16(a, b, acc);
  }
  const int n = tn * 16 + (l & 15);
#pragma unroll
  for (int v = 0; v < 8; ++v) {
    const int m = tm * 16 + v + ((l & 16) ? 8 : 0);
    out[(size_t)m * D_ + n] = acc[v];
  }
}

// ---------------------------------------------------------------------------
extern "C" void kernel_launch(void* const* d_in, const int* in_sizes, int n_in,
                              void* d_out, int out_size, void* d_ws, size_t ws_size,
                              hipStream_t stream) {
  const float* x   = (const float*)d_in[0];
  const float* mem = (const float*)d_in[1];
  const float* pe  = (const float*)d_in[2];
  const float* qw  = (const float*)d_in[3];
  const float* kw  = (const float*)d_in[4];
  const float* vw  = (const float*)d_in[5];
  const float* rw  = (const float*)d_in[6];
  const float* ow  = (const float*)d_in[7];
  const float* bu  = (const float*)d_in[8];
  const float* bv  = (const float*)d_in[9];
  float* out = (float*)d_out;

  // workspace layout (bf16 elements)
  unsigned short* p = (unsigned short*)d_ws;
  const size_t SZ_X = (size_t)B_ * T_ * D_;   // 2M
  const size_t SZ_W = (size_t)D_ * D_;        // 1M
  unsigned short* xb   = p; p += SZ_X;
  unsigned short* memb = p; p += SZ_X;
  unsigned short* peb  = p; p += (size_t)J_ * D_;
  unsigned short* qwb  = p; p += SZ_W;
  unsigned short* kwb  = p; p += SZ_W;
  unsigned short* vwb  = p; p += SZ_W;
  unsigned short* rwb  = p; p += SZ_W;
  unsigned short* owb  = p; p += SZ_W;
  unsigned short* qu   = p; p += (size_t)B_ * H_ * T_ * DH_;
  unsigned short* qv   = p; p += (size_t)B_ * H_ * T_ * DH_;
  unsigned short* kk   = p; p += (size_t)B_ * H_ * L_ * DH_;
  unsigned short* vt   = p; p += (size_t)B_ * H_ * L_ * DH_;   // transposed [b,h,d,L]
  unsigned short* rr   = p; p += (size_t)H_ * J_ * DH_;
  unsigned short* att  = p;                                    // [b,t, h*64+d]

  // one-time f32 -> bf16 conversions
  const int n8x = (int)(SZ_X / 8), n8w = (int)(SZ_W / 8);
  k_cvt<<<(n8x + 255) / 256, 256, 0, stream>>>(x,   xb,   n8x);
  k_cvt<<<(n8x + 255) / 256, 256, 0, stream>>>(mem, memb, n8x);
  k_cvt<<<(n8x + 255) / 256, 256, 0, stream>>>(pe,  peb,  n8x);
  k_cvt<<<(n8w + 255) / 256, 256, 0, stream>>>(qw,  qwb,  n8w);
  k_cvt<<<(n8w + 255) / 256, 256, 0, stream>>>(kw,  kwb,  n8w);
  k_cvt<<<(n8w + 255) / 256, 256, 0, stream>>>(vw,  vwb,  n8w);
  k_cvt<<<(n8w + 255) / 256, 256, 0, stream>>>(rw,  rwb,  n8w);
  k_cvt<<<(n8w + 255) / 256, 256, 0, stream>>>(ow,  owb,  n8w);

  k_proj_q <<<(B_ * T_ / 16) * (D_ / 16) / 8, 256, 0, stream>>>(xb, qwb, bu, bv, qu, qv);
  k_proj_kv<<<(B_ * L_ / 16) * (D_ / 16) / 8, 256, 0, stream>>>(xb, memb, kwb, vwb, kk, vt);
  k_proj_r <<<(J_ / 16) * (D_ / 16) / 8, 256, 0, stream>>>(peb, rwb, rr);

  const size_t smem_bytes =
      (size_t)16 * RELS * 2       // rel_pre bf16
      + 4096                      // wrap row f32
      + 8192                      // P staging
      + 512 + 512                 // m/l stats
      + 8 * 16 * 64 * 4;          // per-wave O partials
  k_attn<<<B_ * H_ * (T_ / 16), 256, smem_bytes, stream>>>(qu, qv, kk, vt, rr, att);

  k_out<<<(B_ * T_ / 16) * (D_ / 16) / 8, 256, 0, stream>>>(att, owb, out);
}
